// TransformerBlock_36670430773491
// MI455X (gfx1250) — matrix operand order repaired and checked by reading, not compiled, not run on hardware
//
#include <hip/hip_runtime.h>

// ---------------------------------------------------------------------------
// MI455X (gfx1250) transformer block, bf16 WMMA path.
//   B=2, T=1024, D=1024, H=64 heads of HS=64 (reference bug), FF=4096.
// All matmuls via v_wmma_f32_16x16x32_bf16 (f32 accumulate).
// - Main GEMM: 128x128 block tile, double-buffered LDS, A tiles staged by the
//   Tensor Data Mover (tensor_load_to_lds, TENSORcnt), one barrier per K-step.
// - Projection/attention: per-lane global_load_async_to_lds_b128 (ASYNCcnt).
// ---------------------------------------------------------------------------

typedef unsigned short u16;
typedef __attribute__((ext_vector_type(16))) __bf16 bf16x16;
typedef __attribute__((ext_vector_type(8)))  float  f32x8;
typedef __attribute__((ext_vector_type(4)))  unsigned uv4;
typedef __attribute__((ext_vector_type(8)))  unsigned uv8;

struct U4 { unsigned int a, b, c, d; };           // POD 16-byte chunk
struct Pack32 { U4 lo, hi; };
union FragU { Pack32 p; bf16x16 f; };
static_assert(sizeof(bf16x16) == 32, "bf16x16 must be 32B");

__device__ inline u16 f32_to_bf16(float f) {
  unsigned u = __float_as_uint(f);
  u += 0x7FFFu + ((u >> 16) & 1u);                // round to nearest even
  return (u16)(u >> 16);
}

// Per-lane async global->LDS copy of 16B (VGLOBAL op 98, tracked by ASYNCcnt).
// LDS operand is the raw LDS byte offset = low 32 bits of the generic pointer.
__device__ inline void async_ld_b128(void* lds_dst, const void* gsrc) {
  unsigned loff = (unsigned)(size_t)lds_dst;
  asm volatile("global_load_async_to_lds_b128 %0, %1, off"
               :: "v"(loff), "v"(gsrc)
               : "memory");
}
__device__ inline void wait_asynccnt0() {
  asm volatile("s_wait_asynccnt 0x0" ::: "memory");
}

// Tensor Data Mover: load a 2D tile (128 rows x 32 bf16, row pitch `stride_el`
// elements) from global into LDS, padding each 64B row with 16B so the LDS
// row stride is 40 u16 (matches our fragment layout). ISA §8.3/8.4 D# fields.
// Issued once per wave (tensor ops ignore EXEC); tracked by TENSORcnt.
__device__ inline void tdm_load_a_tile(const void* gtile, unsigned lds_off,
                                       unsigned stride_el) {
  unsigned long long ga = (unsigned long long)gtile;
  uv4 g0;
  g0[0] = 1u;                                         // count=1 (valid, user)
  g0[1] = lds_off;                                    // lds_addr (bytes)
  g0[2] = (unsigned)(ga & 0xFFFFFFFFu);               // global_addr[31:0]
  g0[3] = (unsigned)((ga >> 32) & 0x01FFFFFFu)        // global_addr[56:32]
        | (2u << 30);                                 // type = 2 ("image")
  uv8 g1;
  g1[0] = (1u << 16)                                  // data_size = 2 bytes
        | (1u << 20)                                  // pad_enable
        | (3u << 22)                                  // pad_interval: 16 DW
        | (3u << 25);                                 // pad_amount: 4 DW
  g1[1] = (32u << 16);                                // tensor_dim0 = 32
  g1[2] = (128u << 16);                               // tensor_dim1 = 128
  g1[3] = (32u << 16);                                // tile_dim0 = 32
  g1[4] = 128u;                                       // tile_dim1 = 128
  g1[5] = stride_el;                                  // tensor_dim0_stride
  g1[6] = 0u;
  g1[7] = 0u;
  asm volatile("tensor_load_to_lds %0, %1" :: "s"(g0), "s"(g1) : "memory");
}

// A-fragment (16x32 bf16, M x K): lane m holds row m; elements i are
// k = kh*16 + i (contiguous 32B run).
__device__ inline bf16x16 load_afrag(const u16* rowbase, int kh) {
  FragU x;
  x.p.lo = *(const U4*)(rowbase + kh * 16);
  x.p.hi = *(const U4*)(rowbase + kh * 16 + 8);
  return x.f;
}
// B-fragment (32x16 bf16, K x N) from an LDS tile stored [n][k]:
// lane n holds column n; elements i = k = kh*16 + i -> contiguous 32B run.
__device__ inline bf16x16 load_bfrag(const u16* rowbase, int kh) {
  FragU x;
  const U4* p = (const U4*)(rowbase + kh * 16);
  x.p.lo = p[0];
  x.p.hi = p[1];
  return x.f;
}

__device__ inline f32x8 wmma_bf16(bf16x16 a, bf16x16 b, f32x8 c) {
  return __builtin_amdgcn_wmma_f32_16x16x32_bf16(false, a, false, b,
                                                 (short)0, c, false, false);
}

// ---------------------------------------------------------------------------
// f32 -> bf16 cast (grid-stride)
// ---------------------------------------------------------------------------
__global__ void cast_f32_bf16(const float* __restrict__ in,
                              u16* __restrict__ out, int n) {
  for (int i = blockIdx.x * blockDim.x + threadIdx.x; i < n;
       i += gridDim.x * blockDim.x)
    out[i] = f32_to_bf16(in[i]);
}

// ---------------------------------------------------------------------------
// Per-head QKV projection: out[b][h][t][e] = sum_d x[b][t][d] * W[h][d][e]
// grid = (T/64, H, B*3); block = 128 (4 waves, each a 32x32 tile of 64x64)
// ---------------------------------------------------------------------------
__global__ __launch_bounds__(128) void proj_qkv(
    const u16* __restrict__ xb,
    const u16* __restrict__ Wq, const u16* __restrict__ Wk,
    const u16* __restrict__ Wv,
    u16* __restrict__ qo, u16* __restrict__ ko, u16* __restrict__ vo,
    int T, int D) {
  const int H = 64, HS = 64;
  const int tb  = blockIdx.x;
  const int h   = blockIdx.y;
  const int b   = blockIdx.z & 1;
  const int sel = blockIdx.z >> 1;

  const u16* W = (sel == 0 ? Wq : sel == 1 ? Wk : Wv) + (size_t)h * D * HS;
  u16* out     = (sel == 0 ? qo : sel == 1 ? ko : vo) +
                 ((size_t)(b * H + h) * T + (size_t)tb * 64) * HS;
  const u16* A = xb + ((size_t)b * T + (size_t)tb * 64) * D;

  __shared__ u16 As[64 * 40];   // [m][k], row stride 40 (80B, 16B aligned)
  __shared__ u16 Bs[64 * 40];   // [n=e][k=d]

  const int tid = threadIdx.x;
  const int w    = tid >> 5;
  const int lane = tid & 31;
  const int m    = lane & 15;
  const int kh   = lane >> 4;
  const int wr   = (w & 1) * 32;
  const int wc   = (w >> 1) * 32;

  f32x8 acc[2][2] = {};

  for (int k0 = 0; k0 < D; k0 += 32) {
    __syncthreads();
    // A tile 64x32, async straight copy
#pragma unroll
    for (int i = 0; i < 2; ++i) {
      int idx = tid + i * 128;                 // 0..255
      int row = idx >> 2, c = (idx & 3) * 8;   // 8 u16 = 16B
      async_ld_b128(&As[row * 40 + c], &A[(size_t)row * D + k0 + c]);
    }
    // W tile 32(d) x 64(e), store transposed [e][d]
#pragma unroll
    for (int i = 0; i < 2; ++i) {
      int idx = tid + i * 128;
      int d = idx >> 3, eg = (idx & 7) * 8;
      U4 t4 = *(const U4*)&W[(size_t)(k0 + d) * HS + eg];
      const u16* t8 = (const u16*)&t4;
#pragma unroll
      for (int j = 0; j < 8; ++j) Bs[(eg + j) * 40 + d] = t8[j];
    }
    wait_asynccnt0();
    __syncthreads();

    bf16x16 a0 = load_afrag(&As[(wr + m) * 40], kh);
    bf16x16 a1 = load_afrag(&As[(wr + 16 + m) * 40], kh);
    bf16x16 b0 = load_bfrag(&Bs[(wc + m) * 40], kh);
    bf16x16 b1 = load_bfrag(&Bs[(wc + 16 + m) * 40], kh);
    acc[0][0] = wmma_bf16(a0, b0, acc[0][0]);
    acc[0][1] = wmma_bf16(a0, b1, acc[0][1]);
    acc[1][0] = wmma_bf16(a1, b0, acc[1][0]);
    acc[1][1] = wmma_bf16(a1, b1, acc[1][1]);
  }

#pragma unroll
  for (int i = 0; i < 2; ++i)
#pragma unroll
    for (int j = 0; j < 2; ++j) {
      int col = wc + j * 16 + m;
#pragma unroll
      for (int r = 0; r < 8; ++r) {
        int row = wr + i * 16 + kh * 8 + r;
        out[(size_t)row * HS + col] = f32_to_bf16(acc[i][j][r]);
      }
    }
}

// ---------------------------------------------------------------------------
// Flash-style causal attention per (b,h), 64 query rows per block.
// Each wave owns 16 query rows; 4 waves cooperate on K/V staging.
// Output written directly into concat layout cat[b][t][e*H + h] as bf16.
// ---------------------------------------------------------------------------
__global__ __launch_bounds__(128) void attn_kernel(
    const u16* __restrict__ q, const u16* __restrict__ k,
    const u16* __restrict__ v, u16* __restrict__ cat, int T) {
  const int H = 64, HS = 64;
  const int qbk = blockIdx.x;
  const int h   = blockIdx.y;
  const int b   = blockIdx.z;
  const int tid = threadIdx.x;
  const int w    = tid >> 5;
  const int lane = tid & 31;
  const int m    = lane & 15;
  const int kh   = lane >> 4;

  __shared__ u16 qs[64 * 72];   // [t][e] row stride 72 (144B, 16B aligned)
  __shared__ u16 ks[64 * 72];   // [s][e]
  __shared__ u16 vs[64 * 72];   // [e][s] (transposed)
  __shared__ u16 ps[64 * 72];   // [t][s] probabilities, wave-private rows

  const size_t hb = (size_t)(b * H + h) * T * HS;
  const u16* qg = q + hb + (size_t)qbk * 64 * HS;
  const u16* kg = k + hb;
  const u16* vg = v + hb;

  // stage Q block once (async; waited inside first loop iteration)
#pragma unroll
  for (int i = 0; i < 2; ++i) {
    int idx = tid + i * 128;
    int row = idx >> 2, c = (idx & 3) * 8;
    async_ld_b128(&qs[row * 72 + c], &qg[(size_t)row * HS + c]);
  }

  f32x8 o[4] = {};
  float mrow[8], lrow[8];
#pragma unroll
  for (int r = 0; r < 8; ++r) { mrow[r] = -3.0e38f; lrow[r] = 0.f; }

  for (int kb = 0; kb <= qbk; ++kb) {
    __syncthreads();   // previous iter's ks/vs reads done
#pragma unroll
    for (int i = 0; i < 2; ++i) {      // K block, async straight copy [s][e]
      int idx = tid + i * 128;
      int row = idx >> 2, c = (idx & 3) * 8;
      async_ld_b128(&ks[row * 72 + c],
                    &kg[(size_t)(kb * 64 + row) * HS + c]);
    }
#pragma unroll
    for (int i = 0; i < 4; ++i) {      // V block, transposed to [e][s]
      int idx = tid + i * 128;
      int s2 = idx >> 3, eg = (idx & 7) * 8;
      U4 t4 = *(const U4*)&vg[(size_t)(kb * 64 + s2) * HS + eg];
      const u16* t8 = (const u16*)&t4;
#pragma unroll
      for (int j = 0; j < 8; ++j) vs[(eg + j) * 72 + s2] = t8[j];
    }
    wait_asynccnt0();
    __syncthreads();

    // S = q @ k^T : A = q[t][e], B(lds [n=s][k=e]) = ks row-major directly
    f32x8 sacc[4] = {};
#pragma unroll
    for (int es = 0; es < 64; es += 32) {
      bf16x16 a = load_afrag(&qs[(w * 16 + m) * 72 + es], kh);
#pragma unroll
      for (int nt = 0; nt < 4; ++nt) {
        bf16x16 bfr = load_bfrag(&ks[(nt * 16 + m) * 72 + es], kh);
        sacc[nt] = wmma_bf16(a, bfr, sacc[nt]);
      }
    }

    // causal mask + online softmax; lane holds col m of rows rbase..rbase+7
    const int rbase = qbk * 64 + w * 16 + kh * 8;
#pragma unroll
    for (int r = 0; r < 8; ++r) {
      float mx = -3.0e38f;
#pragma unroll
      for (int nt = 0; nt < 4; ++nt) {
        int col = kb * 64 + nt * 16 + m;
        float sv = sacc[nt][r] * 0.125f;          // 1/sqrt(64)
        if (col > rbase + r) sv = -3.0e38f;
        sacc[nt][r] = sv;
        mx = fmaxf(mx, sv);
      }
#pragma unroll
      for (int off = 8; off >= 1; off >>= 1)      // reduce over 16-lane half
        mx = fmaxf(mx, __shfl_xor(mx, off, 32));
      float mnew  = fmaxf(mrow[r], mx);
      float alpha = __expf(mrow[r] - mnew);
      float rsum = 0.f;
#pragma unroll
      for (int nt = 0; nt < 4; ++nt) {
        float pv = __expf(sacc[nt][r] - mnew);
        sacc[nt][r] = pv;
        rsum += pv;
      }
#pragma unroll
      for (int off = 8; off >= 1; off >>= 1)
        rsum += __shfl_xor(rsum, off, 32);
      lrow[r] = lrow[r] * alpha + rsum;
      mrow[r] = mnew;
#pragma unroll
      for (int et = 0; et < 4; ++et) o[et][r] *= alpha;
    }

    // spill P (C layout) to wave-private LDS rows, reload as A fragments.
    // Same-wave DS ops are in order -> no barrier needed.
#pragma unroll
    for (int nt = 0; nt < 4; ++nt)
#pragma unroll
      for (int r = 0; r < 8; ++r)
        ps[(w * 16 + kh * 8 + r) * 72 + nt * 16 + m] = f32_to_bf16(sacc[nt][r]);

    // O += P @ V : A = ps[t][s], B(lds [n=e][k=s]) = vs
#pragma unroll
    for (int ss = 0; ss < 64; ss += 32) {
      bf16x16 a = load_afrag(&ps[(w * 16 + m) * 72 + ss], kh);
#pragma unroll
      for (int et = 0; et < 4; ++et) {
        bf16x16 bfr = load_bfrag(&vs[(et * 16 + m) * 72 + ss], kh);
        o[et] = wmma_bf16(a, bfr, o[et]);
      }
    }
  }

  // normalize and write to concat layout: cat[((b*T+t)*HS + e)*H + h]
  const int tbase = qbk * 64 + w * 16 + kh * 8;
#pragma unroll
  for (int et = 0; et < 4; ++et) {
    int e = et * 16 + m;
#pragma unroll
    for (int r = 0; r < 8; ++r) {
      float val = o[et][r] / lrow[r];
      size_t ci = ((size_t)(b * T + tbase + r) * HS + e) * H + h;
      cat[ci] = f32_to_bf16(val);
    }
  }
}

// ---------------------------------------------------------------------------
// Generic bf16 GEMM: C = A(MxK) @ B(KxN) + bias, optional ReLU.
// 256 threads = 8 waves; block tile 128x128; wave tile 32x64 (2x4 WMMAs).
// Double-buffered LDS; A tiles staged by TDM (wave 0), B tiles by all waves
// (transposed store). One barrier per K-step. grid = (N/128, M/128).
// ---------------------------------------------------------------------------
template <bool RELU>
__global__ __launch_bounds__(256) void gemm_bf16_kernel(
    const u16* __restrict__ A, const u16* __restrict__ Bm,
    const float* __restrict__ bias,
    float* __restrict__ Cf, u16* __restrict__ Cb, int M, int N, int K) {
  const int nb = blockIdx.x;
  const int mb = blockIdx.y;

  __shared__ u16 As[2][128 * 40];   // [m][k] row stride 40
  __shared__ u16 Bs[2][128 * 40];   // [n][k]

  const int tid  = threadIdx.x;
  const int w    = tid >> 5;
  const int lane = tid & 31;
  const int m    = lane & 15;
  const int kh   = lane >> 4;
  const int wr   = (w & 3) * 32;    // 4 row groups
  const int wc   = (w >> 2) * 64;   // 2 col groups

  f32x8 acc[2][4] = {};

  auto stage = [&](int bi2, int k0) {
    // A tile 128x32: one TDM descriptor, issued by wave 0 only.
    if (w == 0)
      tdm_load_a_tile(&A[(size_t)(mb * 128) * K + k0],
                      (unsigned)(size_t)&As[bi2][0], (unsigned)K);
    // B tile 32(k) x 128(n), store transposed [n][k]
#pragma unroll
    for (int i = 0; i < 2; ++i) {
      int idx = tid + i * 256;
      int d = idx >> 4, eg = (idx & 15) * 8;
      U4 t4 = *(const U4*)&Bm[(size_t)(k0 + d) * N + nb * 128 + eg];
      const u16* t8 = (const u16*)&t4;
#pragma unroll
      for (int j2 = 0; j2 < 8; ++j2) Bs[bi2][(eg + j2) * 40 + d] = t8[j2];
    }
  };

  stage(0, 0);
  int bi = 0;
  for (int k0 = 0; k0 < K; k0 += 32, bi ^= 1) {
    if (w == 0) __builtin_amdgcn_s_wait_tensorcnt(0);  // A tile landed
    __syncthreads();   // releases only after wave 0 waited; prev reads done
    if (k0 + 32 < K) stage(bi ^ 1, k0 + 32);           // prefetch next tile
    if (k0 + 64 < K)                                   // hint one further
      __builtin_prefetch(&Bm[(size_t)(k0 + 64) * N + nb * 128], 0, 1);

    bf16x16 a0 = load_afrag(&As[bi][(wr + m) * 40], kh);
    bf16x16 a1 = load_afrag(&As[bi][(wr + 16 + m) * 40], kh);
#pragma unroll
    for (int j = 0; j < 4; ++j) {
      bf16x16 bj = load_bfrag(&Bs[bi][(wc + j * 16 + m) * 40], kh);
      acc[0][j] = wmma_bf16(a0, bj, acc[0][j]);
      acc[1][j] = wmma_bf16(a1, bj, acc[1][j]);
    }
  }

#pragma unroll
  for (int i = 0; i < 2; ++i)
#pragma unroll
    for (int j = 0; j < 4; ++j) {
      int col = nb * 128 + wc + j * 16 + m;
      float bv = bias ? bias[col] : 0.0f;
#pragma unroll
      for (int r = 0; r < 8; ++r) {
        int row = mb * 128 + wr + i * 16 + kh * 8 + r;
        float vv = acc[i][j][r] + bv;
        if (RELU) vv = fmaxf(vv, 0.0f);
        size_t idx = (size_t)row * N + col;
        if (Cf) Cf[idx] = vv;
        if (Cb) Cb[idx] = f32_to_bf16(vv);
      }
    }
}

// ---------------------------------------------------------------------------
// out = LayerNorm(xa + xb) * g + bt ; optional bf16 copy. One block per row.
// ---------------------------------------------------------------------------
__global__ __launch_bounds__(256) void add_ln(
    const float* __restrict__ xa, const float* __restrict__ xb2,
    const float* __restrict__ g, const float* __restrict__ bt,
    float* __restrict__ outf, u16* __restrict__ outb, int D) {
  const int row = blockIdx.x;
  const int tid = threadIdx.x;
  __shared__ float red[8];

  float vals[4];
  float s = 0.f;
#pragma unroll
  for (int i = 0; i < 4; ++i) {
    int c = tid + i * 256;
    float v = xa[(size_t)row * D + c] + xb2[(size_t)row * D + c];
    vals[i] = v;
    s += v;
  }
#pragma unroll
  for (int off = 16; off >= 1; off >>= 1) s += __shfl_xor(s, off, 32);
  if ((tid & 31) == 0) red[tid >> 5] = s;
  __syncthreads();
  float tot = 0.f;
#pragma unroll
  for (int i = 0; i < 8; ++i) tot += red[i];
  const float mu = tot / (float)D;
  __syncthreads();

  float vsum = 0.f;
#pragma unroll
  for (int i = 0; i < 4; ++i) { float d2 = vals[i] - mu; vsum += d2 * d2; }
#pragma unroll
  for (int off = 16; off >= 1; off >>= 1) vsum += __shfl_xor(vsum, off, 32);
  if ((tid & 31) == 0) red[tid >> 5] = vsum;
  __syncthreads();
  float vtot = 0.f;
#pragma unroll
  for (int i = 0; i < 8; ++i) vtot += red[i];
  const float rstd = rsqrtf(vtot / (float)D + 1e-5f);

#pragma unroll
  for (int i = 0; i < 4; ++i) {
    int c = tid + i * 256;
    float ov = (vals[i] - mu) * rstd * g[c] + bt[c];
    size_t idx = (size_t)row * D + c;
    if (outf) outf[idx] = ov;
    if (outb) outb[idx] = f32_to_bf16(ov);
  }
}

// ---------------------------------------------------------------------------
extern "C" void kernel_launch(void* const* d_in, const int* in_sizes, int n_in,
                              void* d_out, int out_size, void* d_ws,
                              size_t ws_size, hipStream_t stream) {
  const int B = 2, T = 1024, D = 1024, H = 64, HS = 64, FF = 4096;
  const float* x   = (const float*)d_in[0];
  const float* Wq  = (const float*)d_in[1];
  const float* Wk  = (const float*)d_in[2];
  const float* Wv  = (const float*)d_in[3];
  const float* Wo  = (const float*)d_in[4];
  const float* bo  = (const float*)d_in[5];
  const float* W1  = (const float*)d_in[6];
  const float* b1  = (const float*)d_in[7];
  const float* W2  = (const float*)d_in[8];
  const float* b2  = (const float*)d_in[9];
  const float* g1  = (const float*)d_in[10];
  const float* be1 = (const float*)d_in[11];
  const float* g2  = (const float*)d_in[12];
  const float* be2 = (const float*)d_in[13];

  char* p = (char*)d_ws;
  auto alloc = [&](size_t bytes) -> void* {
    void* r = (void*)p;
    p += (bytes + 255) & ~(size_t)255;
    return r;
  };
  const size_t nx   = (size_t)B * T * D;       // 2M
  const size_t nW   = (size_t)H * D * HS;      // 4M
  const size_t nqkv = (size_t)B * H * T * HS;  // 8M
  const size_t ncat = (size_t)B * T * H * HS;  // 8M

  u16* xb   = (u16*)alloc(nx * 2);
  u16* Wqb  = (u16*)alloc(nW * 2);
  u16* Wkb  = (u16*)alloc(nW * 2);
  u16* Wvb  = (u16*)alloc(nW * 2);
  u16* Wob  = (u16*)alloc((size_t)H * HS * D * 2);
  u16* W1b  = (u16*)alloc((size_t)D * FF * 2);
  u16* W2b  = (u16*)alloc((size_t)FF * D * 2);
  u16* qb   = (u16*)alloc(nqkv * 2);
  u16* kb   = (u16*)alloc(nqkv * 2);
  u16* vb   = (u16*)alloc(nqkv * 2);
  u16* catb = (u16*)alloc(ncat * 2);
  float* mha = (float*)alloc(nx * 4);
  float* x1  = (float*)alloc(nx * 4);
  u16*   x1b = (u16*)alloc(nx * 2);
  u16*   h1b = (u16*)alloc((size_t)B * T * FF * 2);
  float* ff  = (float*)alloc(nx * 4);

  auto cast = [&](const float* src, u16* dst, size_t n) {
    cast_f32_bf16<<<2048, 256, 0, stream>>>(src, dst, (int)n);
  };
  cast(x, xb, nx);
  cast(Wq, Wqb, nW);
  cast(Wk, Wkb, nW);
  cast(Wv, Wvb, nW);
  cast(Wo, Wob, (size_t)H * HS * D);
  cast(W1, W1b, (size_t)D * FF);
  cast(W2, W2b, (size_t)FF * D);

  proj_qkv<<<dim3(T / 64, H, B * 3), 128, 0, stream>>>(xb, Wqb, Wkb, Wvb,
                                                       qb, kb, vb, T, D);

  attn_kernel<<<dim3(T / 64, H, B), 128, 0, stream>>>(qb, kb, vb, catb, T);

  // mha = cat @ Wo + bo                (M=2048, N=1024, K=4096) -> f32
  gemm_bf16_kernel<false><<<dim3(D / 128, (B * T) / 128), 256, 0, stream>>>(
      catb, Wob, bo, mha, nullptr, B * T, D, H * HS);

  // x1 = LN(x + mha)                   -> f32 + bf16
  add_ln<<<B * T, 256, 0, stream>>>(x, mha, g1, be1, x1, x1b, D);

  // h = relu(x1 @ W1 + b1)             (M=2048, N=4096, K=1024) -> bf16
  gemm_bf16_kernel<true><<<dim3(FF / 128, (B * T) / 128), 256, 0, stream>>>(
      x1b, W1b, b1, nullptr, h1b, B * T, FF, D);

  // ff = h @ W2 + b2                   (M=2048, N=1024, K=4096) -> f32
  gemm_bf16_kernel<false><<<dim3(D / 128, (B * T) / 128), 256, 0, stream>>>(
      h1b, W2b, b2, ff, nullptr, B * T, D, FF);

  // out = LN(x1 + ff)                  -> f32 (d_out)
  add_ln<<<B * T, 256, 0, stream>>>(x1, ff, g2, be2, (float*)d_out, nullptr, D);
}
